// ROIHead_27736898798119
// MI455X (gfx1250) — matrix operand order repaired and checked by reading, not compiled, not run on hardware
//
#include <hip/hip_runtime.h>

// ---------------- problem constants ----------------
#define BATCH   2
#define NPROP   1000
#define CHN     256
#define HFEAT   200
#define POOL    7
#define KDIM    (CHN * POOL * POOL)   // 12544
#define HID     1024
#define NCLS    91
#define NREG    (NCLS * 4)            // 364
#define NHEAD   464                   // 91 + 364 padded to multiple of 16
#define MROWS   (BATCH * NPROP)       // 2000
#define MCAND   (NPROP * NCLS)        // 91000 per batch
#define K_POST  100
#define NEGV    (-1.0e9f)
#define IMGSZ   800.0f
#define BBOX_CLAMP 4.135166556742356f // log(1000/16)

typedef __attribute__((ext_vector_type(16))) __bf16    v16bf;
typedef __attribute__((ext_vector_type(8)))  float     v8f;
typedef __attribute__((ext_vector_type(4)))  unsigned  u32x4;

__device__ __forceinline__ unsigned short f2bf(float f) {
    unsigned u = __float_as_uint(f);
    unsigned r = u + 0x7FFFu + ((u >> 16) & 1u);   // round-to-nearest-even
    return (unsigned short)(r >> 16);
}

// gfx1250 async global->LDS copy, 16B per lane, tracked by ASYNCcnt.
__device__ __forceinline__ void async_b128(unsigned lds_addr, const void* gptr) {
    asm volatile("global_load_async_to_lds_b128 %0, %1, off"
                 :: "v"(lds_addr), "v"((unsigned long long)(uintptr_t)gptr)
                 : "memory");
}
__device__ __forceinline__ void wait_async0() {
    asm volatile("s_wait_asynccnt 0x0" ::: "memory");
}
__device__ __forceinline__ void wait_async6() {   // drain previous stage only
    asm volatile("s_wait_asynccnt 0x6" ::: "memory");
}

// ---------------- fp32 -> bf16 transposed convert (W[K][N] -> WT[N][K]) ---
#define TRT 32
__global__ __launch_bounds__(1024)
void cvt_bf16_transpose_kernel(const float* __restrict__ W,
                               unsigned short* __restrict__ WT,
                               int K, int N) {
    __shared__ float tile[TRT][TRT + 1];
    int k0 = blockIdx.x * TRT;
    int n0 = blockIdx.y * TRT;
    int tx = threadIdx.x & 31, ty = threadIdx.x >> 5;
    int k = k0 + ty, n = n0 + tx;
    tile[ty][tx] = (k < K && n < N) ? W[(long)k * N + n] : 0.f;
    __syncthreads();
    int kt = k0 + tx, nt = n0 + ty;
    if (nt < N && kt < K)
        WT[(long)nt * K + kt] = f2bf(tile[tx][ty]);
}

// pack Wcls (1024x91) + Wreg (1024x364) transposed into WhT (464 x 1024) bf16,
// plus padded 464-float bias.
__global__ void head_pack_T_kernel(const float* __restrict__ Wcls,
                                   const float* __restrict__ bcls,
                                   const float* __restrict__ Wreg,
                                   const float* __restrict__ breg,
                                   unsigned short* __restrict__ WhT,
                                   float* __restrict__ bh) {
    long i = (long)blockIdx.x * blockDim.x + threadIdx.x;
    long total = (long)NHEAD * HID;
    if (i >= total) return;
    int j = (int)(i / HID);   // head column (row of WT)
    int k = (int)(i % HID);
    float v = 0.f;
    if (j < NCLS)             v = Wcls[(long)k * NCLS + j];
    else if (j < NCLS + NREG) v = Wreg[(long)k * NREG + (j - NCLS)];
    WhT[i] = f2bf(v);
    if (k == 0) {
        float bv = 0.f;
        if (j < NCLS)             bv = bcls[j];
        else if (j < NCLS + NREG) bv = breg[j - NCLS];
        bh[j] = bv;
    }
}

// ---------------- ROI-align, writes A matrix (2000 x 12544) in bf16 -------
__global__ void roi_align_kernel(const float* __restrict__ proposals,
                                 const float* __restrict__ features,
                                 unsigned short* __restrict__ A) {
    long gid = (long)blockIdx.x * blockDim.x + threadIdx.x;
    const long total = (long)MROWS * KDIM;
    if (gid >= total) return;
    int col = (int)(gid % KDIM);
    int row = (int)(gid / KDIM);
    int n = row % NPROP, b = row / NPROP;
    int c = col / (POOL * POOL);
    int p = col % (POOL * POOL);
    int py = p / POOL, px = p % POOL;

    const float* pr = proposals + ((long)b * NPROP + n) * 4;
    float x1 = pr[0] * 0.25f, y1 = pr[1] * 0.25f;
    float x2 = pr[2] * 0.25f, y2 = pr[3] * 0.25f;
    float gy = (py + 0.5f) / (float)POOL;
    float gx = (px + 0.5f) / (float)POOL;
    float yy = y1 + gy * (y2 - y1);
    float xx = x1 + gx * (x2 - x1);
    yy = fminf(fmaxf(yy, 0.f), (float)(HFEAT - 1));
    xx = fminf(fmaxf(xx, 0.f), (float)(HFEAT - 1));
    float y0f = floorf(yy), x0f = floorf(xx);
    float wy = yy - y0f, wx = xx - x0f;
    int y0 = (int)y0f, x0 = (int)x0f;
    int y1i = min(y0 + 1, HFEAT - 1), x1i = min(x0 + 1, HFEAT - 1);
    const float* f = features + (((long)b * CHN + c) * HFEAT) * HFEAT;
    float f00 = f[y0  * HFEAT + x0 ], f01 = f[y0  * HFEAT + x1i];
    float f10 = f[y1i * HFEAT + x0 ], f11 = f[y1i * HFEAT + x1i];
    float val = f00 * (1.f - wy) * (1.f - wx) + f01 * (1.f - wy) * wx
              + f10 *        wy * (1.f - wx) + f11 *        wy * wx;
    A[gid] = f2bf(val);
}

// ---------------- bf16 WMMA GEMM: C = act(A @ BT^T + bias) ----------------
// A row-major [M][K], BT row-major [N][K]. Block tile 128x64, K-step 64.
// 8 waves, 32x32 register tile per wave (4 wmma accumulators).
// Double-buffered async global->LDS staging, unrolled x2 so buffer indices
// are compile-time constants. REQUIRES: K % 128 == 0 (12544, 1024 both ok).
#define TM 128
#define TN 64
#define TKS 64
#define LSTR 72   // ushorts per LDS row (144B, 16B aligned)

__global__ __launch_bounds__(256)
void gemm_bf16_wmma_kernel(const unsigned short* __restrict__ A,  int lda,
                           const unsigned short* __restrict__ BT, int ldbt,
                           const float* __restrict__ bias,
                           void* __restrict__ C, int ldc,
                           int M, int N, int K, int relu, int out_bf16) {
    __shared__ unsigned short As[2][TM * LSTR];   // 2 x 18432 B
    __shared__ unsigned short Bs[2][TN * LSTR];   // 2 x  9216 B

    int tid  = threadIdx.x;
    int lane = tid & 31;
    int wave = tid >> 5;
    int wr = wave & 3;          // 4 wave rows of 32
    int wc = wave >> 2;         // 2 wave cols of 32
    int m0 = blockIdx.x * TM;
    int n0 = blockIdx.y * TN;
    int khalf = lane >> 4;
    int l15   = lane & 15;

    // --- staging assignments: A = 1024 16B-chunks (4/thread), B = 512 (2/thread)
    const unsigned short* pA[4]; unsigned ldA[2][4];
    #pragma unroll
    for (int t = 0; t < 4; ++t) {
        int c = tid + t * 256;
        int r = c >> 3, cc = (c & 7) * 8;
        int ma = min(m0 + r, M - 1);        // clamp: padded rows never stored
        pA[t] = A + (long)ma * lda + cc;
        ldA[0][t] = (unsigned)(uintptr_t)&As[0][r * LSTR + cc];
        ldA[1][t] = (unsigned)(uintptr_t)&As[1][r * LSTR + cc];
    }
    const unsigned short* pB[2]; unsigned ldB[2][2];
    #pragma unroll
    for (int t = 0; t < 2; ++t) {
        int c = tid + t * 256;
        int r = c >> 3, cc = (c & 7) * 8;
        int nb = min(n0 + r, N - 1);        // clamp: padded cols never stored
        pB[t] = BT + (long)nb * ldbt + cc;
        ldB[0][t] = (unsigned)(uintptr_t)&Bs[0][r * LSTR + cc];
        ldB[1][t] = (unsigned)(uintptr_t)&Bs[1][r * LSTR + cc];
    }

    v8f acc[2][2] = {};
    int arow0 = wr * 32 + l15;
    int bn0   = wc * 32 + l15;

    // fragment assembly: every half-fragment is one aligned 16B LDS vector
    union Frag { v16bf v; u32x4 q[2]; };
    auto compute_stage = [&](const unsigned short* as, const unsigned short* bs) {
        #pragma unroll
        for (int s = 0; s < 2; ++s) {
            Frag af0, af1, bf0, bf1;
            int ka = s * 32 + khalf * 8;          // A: dwords 0-3 at ka, 4-7 at ka+16
            af0.q[0] = *(const u32x4*)&as[ arow0       * LSTR + ka];
            af0.q[1] = *(const u32x4*)&as[ arow0       * LSTR + ka + 16];
            af1.q[0] = *(const u32x4*)&as[(arow0 + 16) * LSTR + ka];
            af1.q[1] = *(const u32x4*)&as[(arow0 + 16) * LSTR + ka + 16];
            int kb = s * 32 + khalf * 8;          // B: dwords 0-3 at kb, 4-7 at kb+16
            bf0.q[0] = *(const u32x4*)&bs[ bn0        * LSTR + kb];
            bf0.q[1] = *(const u32x4*)&bs[ bn0        * LSTR + kb + 16];
            bf1.q[0] = *(const u32x4*)&bs[(bn0 + 16)  * LSTR + kb];
            bf1.q[1] = *(const u32x4*)&bs[(bn0 + 16)  * LSTR + kb + 16];
            acc[0][0] = __builtin_amdgcn_wmma_f32_16x16x32_bf16(
                false, af0.v, false, bf0.v, (short)0, acc[0][0], false, false);
            acc[0][1] = __builtin_amdgcn_wmma_f32_16x16x32_bf16(
                false, af0.v, false, bf1.v, (short)0, acc[0][1], false, false);
            acc[1][0] = __builtin_amdgcn_wmma_f32_16x16x32_bf16(
                false, af1.v, false, bf0.v, (short)0, acc[1][0], false, false);
            acc[1][1] = __builtin_amdgcn_wmma_f32_16x16x32_bf16(
                false, af1.v, false, bf1.v, (short)0, acc[1][1], false, false);
        }
    };

    // prologue: stage 0 into buffer 0
    #pragma unroll
    for (int t = 0; t < 4; ++t) async_b128(ldA[0][t], pA[t]);
    #pragma unroll
    for (int t = 0; t < 2; ++t) async_b128(ldB[0][t], pB[t]);

    for (int k0 = 0; k0 < K; k0 += 2 * TKS) {
        // --- phase 0: compute buf0 (holds k0), prefetch k0+64 -> buf1
        #pragma unroll
        for (int t = 0; t < 4; ++t) async_b128(ldA[1][t], pA[t] + TKS);
        #pragma unroll
        for (int t = 0; t < 2; ++t) async_b128(ldB[1][t], pB[t] + TKS);
        wait_async6();
        __syncthreads();
        compute_stage(As[0], Bs[0]);
        __syncthreads();

        // --- phase 1: compute buf1 (holds k0+64), prefetch k0+128 -> buf0
        #pragma unroll
        for (int t = 0; t < 4; ++t) pA[t] += 2 * TKS;
        #pragma unroll
        for (int t = 0; t < 2; ++t) pB[t] += 2 * TKS;
        if (k0 + 2 * TKS < K) {
            #pragma unroll
            for (int t = 0; t < 4; ++t) async_b128(ldA[0][t], pA[t]);
            #pragma unroll
            for (int t = 0; t < 2; ++t) async_b128(ldB[0][t], pB[t]);
            wait_async6();
        } else {
            wait_async0();
        }
        __syncthreads();
        compute_stage(As[1], Bs[1]);
        __syncthreads();
    }

    // store (ISA C/D layout: lanes 0-15 -> M=r, lanes 16-31 -> M=r+8)
    #pragma unroll
    for (int sn = 0; sn < 2; ++sn) {
        int n = n0 + wc * 32 + sn * 16 + l15;
        bool okN = n < N;
        float bv = okN ? bias[n] : 0.f;
        #pragma unroll
        for (int sm = 0; sm < 2; ++sm) {
            v8f a = acc[sm][sn];
            #pragma unroll
            for (int r = 0; r < 8; ++r) {
                int m = m0 + wr * 32 + sm * 16 + r + khalf * 8;
                if (okN && m < M) {
                    float x = a[r] + bv;
                    if (relu) x = fmaxf(x, 0.f);
                    if (out_bf16) ((unsigned short*)C)[(long)m * ldc + n] = f2bf(x);
                    else          ((float*)C)[(long)m * ldc + n] = x;
                }
            }
        }
    }
}

// ---------------- softmax + box decode + validity -------------------------
__global__ __launch_bounds__(128)
void prep_kernel(const float* __restrict__ proposals,
                 const float* __restrict__ Chead,
                 float* __restrict__ boxes, float* __restrict__ scores,
                 float* __restrict__ s0, float* __restrict__ areas) {
    int i = blockIdx.x;                 // roi index 0..1999
    int b = i / NPROP, n = i % NPROP;
    int j = threadIdx.x;                // class lane
    __shared__ float red[128];
    const float* row = Chead + (long)i * NHEAD;

    float logit = (j < NCLS) ? row[j] : -INFINITY;
    red[j] = logit; __syncthreads();
    for (int s = 64; s > 0; s >>= 1) {
        if (j < s) red[j] = fmaxf(red[j], red[j + s]);
        __syncthreads();
    }
    float mx = red[0]; __syncthreads();
    float e = (j < NCLS) ? __expf(logit - mx) : 0.f;
    red[j] = e; __syncthreads();
    for (int s = 64; s > 0; s >>= 1) {
        if (j < s) red[j] += red[j + s];
        __syncthreads();
    }
    float sum = red[0];
    if (j >= NCLS) return;
    float sc = e / sum;

    const float* pr = proposals + ((long)b * NPROP + n) * 4;
    float w = pr[2] - pr[0], h = pr[3] - pr[1];
    float cx = pr[0] + 0.5f * w, cy = pr[1] + 0.5f * h;
    const float* d = row + NCLS + j * 4;
    float dx = d[0] * 0.1f, dy = d[1] * 0.1f;
    float dw = fminf(d[2] * 0.2f, BBOX_CLAMP);
    float dh = fminf(d[3] * 0.2f, BBOX_CLAMP);
    float pcx = dx * w + cx, pcy = dy * h + cy;
    float pw = __expf(dw) * w, ph = __expf(dh) * h;
    float bx1 = fminf(fmaxf(pcx - 0.5f * pw, 0.f), IMGSZ);
    float by1 = fminf(fmaxf(pcy - 0.5f * ph, 0.f), IMGSZ);
    float bx2 = fminf(fmaxf(pcx + 0.5f * pw, 0.f), IMGSZ);
    float by2 = fminf(fmaxf(pcy + 0.5f * ph, 0.f), IMGSZ);

    long m = (long)b * MCAND + n * NCLS + j;
    boxes[m * 4 + 0] = bx1; boxes[m * 4 + 1] = by1;
    boxes[m * 4 + 2] = bx2; boxes[m * 4 + 3] = by2;
    float bw = bx2 - bx1, bh = by2 - by1;
    scores[m] = sc;
    areas[m]  = bw * bh;
    bool valid = (sc > 0.01f) && (bw >= 0.01f) && (bh >= 0.01f) && (j > 0);
    s0[m] = valid ? sc : NEGV;
}

// ---------------- iterative NMS (one workgroup per batch) -----------------
__global__ __launch_bounds__(1024)
void nms_kernel(const float* __restrict__ boxes, const float* __restrict__ scores,
                const float* __restrict__ s0, const float* __restrict__ areas,
                float* __restrict__ scur, float* __restrict__ out) {
    int b = blockIdx.x;
    int tid = threadIdx.x;
    const float* bx  = boxes  + (long)b * MCAND * 4;
    const float* sc  = scores + (long)b * MCAND;
    const float* s0b = s0     + (long)b * MCAND;
    const float* ar  = areas  + (long)b * MCAND;
    float* s = scur + (long)b * MCAND;
    for (int m = tid; m < MCAND; m += 1024) s[m] = s0b[m];
    __syncthreads();

    __shared__ float rv[1024];
    __shared__ int   ri[1024];
    __shared__ float bbox[5];

    for (int it = 0; it < K_POST; ++it) {
        float best = -INFINITY; int besti = 0;
        for (int m = tid; m < MCAND; m += 1024) {
            float v = s[m];
            if (v > best) { best = v; besti = m; }
        }
        rv[tid] = best; ri[tid] = besti; __syncthreads();
        for (int ss = 512; ss > 0; ss >>= 1) {
            if (tid < ss) {
                float v2 = rv[tid + ss]; int i2 = ri[tid + ss];
                if (v2 > rv[tid] || (v2 == rv[tid] && i2 < ri[tid])) {
                    rv[tid] = v2; ri[tid] = i2;
                }
            }
            __syncthreads();
        }
        if (tid == 0) {
            int ib = ri[0];
            bool ok = rv[0] > NEGV * 0.5f;
            bbox[0] = bx[ib * 4 + 0]; bbox[1] = bx[ib * 4 + 1];
            bbox[2] = bx[ib * 4 + 2]; bbox[3] = bx[ib * 4 + 3];
            bbox[4] = ar[ib];
            float* ob = out + (long)b * K_POST * 4 + it * 4;
            float* os = out + BATCH * K_POST * 4 + (long)b * K_POST + it;
            float* ol = out + BATCH * K_POST * 5 + (long)b * K_POST + it;
            if (ok) {
                ob[0] = bbox[0]; ob[1] = bbox[1]; ob[2] = bbox[2]; ob[3] = bbox[3];
                *os = sc[ib];
                *ol = (float)(ib % NCLS);
            } else {
                ob[0] = ob[1] = ob[2] = ob[3] = 0.f; *os = 0.f; *ol = 0.f;
            }
        }
        __syncthreads();
        float x1 = bbox[0], y1 = bbox[1], x2 = bbox[2], y2 = bbox[3], ba = bbox[4];
        for (int m = tid; m < MCAND; m += 1024) {
            float xx1 = fmaxf(bx[m * 4 + 0], x1);
            float yy1 = fmaxf(bx[m * 4 + 1], y1);
            float xx2 = fminf(bx[m * 4 + 2], x2);
            float yy2 = fminf(bx[m * 4 + 3], y2);
            float inter = fmaxf(xx2 - xx1, 0.f) * fmaxf(yy2 - yy1, 0.f);
            float iou = inter / (ar[m] + ba - inter + 1e-9f);
            if (iou > 0.5f) s[m] = NEGV;
        }
        __syncthreads();
    }
}

// ---------------- host-side launcher ----------------
extern "C" void kernel_launch(void* const* d_in, const int* in_sizes, int n_in,
                              void* d_out, int out_size, void* d_ws, size_t ws_size,
                              hipStream_t stream) {
    const float* proposals = (const float*)d_in[0];
    const float* features  = (const float*)d_in[1];
    const float* W1   = (const float*)d_in[2];
    const float* b1   = (const float*)d_in[3];
    const float* W2   = (const float*)d_in[4];
    const float* b2   = (const float*)d_in[5];
    const float* Wcls = (const float*)d_in[6];
    const float* bcls = (const float*)d_in[7];
    const float* Wreg = (const float*)d_in[8];
    const float* breg = (const float*)d_in[9];
    float* out = (float*)d_out;

    char* ws = (char*)d_ws;
    size_t off = 0;
    auto take = [&](size_t bytes) { char* p = ws + off; off = (off + bytes + 255) & ~(size_t)255; return p; };

    unsigned short* Abf  = (unsigned short*)take((size_t)MROWS * KDIM * 2);
    unsigned short* W1T  = (unsigned short*)take((size_t)KDIM * HID * 2);   // [1024][12544]
    unsigned short* W2T  = (unsigned short*)take((size_t)HID * HID * 2);    // [1024][1024]
    unsigned short* WhT  = (unsigned short*)take((size_t)NHEAD * HID * 2);  // [464][1024]
    float*          bh   = (float*)take(NHEAD * 4);
    unsigned short* H1   = (unsigned short*)take((size_t)MROWS * HID * 2);
    unsigned short* H2   = (unsigned short*)take((size_t)MROWS * HID * 2);
    float*          Ch   = (float*)take((size_t)MROWS * NHEAD * 4);
    float* boxesW  = (float*)take((size_t)BATCH * MCAND * 4 * 4);
    float* scoresW = (float*)take((size_t)BATCH * MCAND * 4);
    float* s0W     = (float*)take((size_t)BATCH * MCAND * 4);
    float* areasW  = (float*)take((size_t)BATCH * MCAND * 4);
    float* scurW   = (float*)take((size_t)BATCH * MCAND * 4);
    (void)ws_size; (void)in_sizes; (void)n_in; (void)out_size;

    // transposed bf16 weight conversion
    dim3 gt1(KDIM / TRT, HID / TRT);
    cvt_bf16_transpose_kernel<<<gt1, 1024, 0, stream>>>(W1, W1T, KDIM, HID);
    dim3 gt2(HID / TRT, HID / TRT);
    cvt_bf16_transpose_kernel<<<gt2, 1024, 0, stream>>>(W2, W2T, HID, HID);
    long nh = (long)NHEAD * HID;
    head_pack_T_kernel<<<(unsigned)((nh + 255) / 256), 256, 0, stream>>>(
        Wcls, bcls, Wreg, breg, WhT, bh);

    // ROI align -> bf16 A matrix
    long ntot = (long)MROWS * KDIM;
    roi_align_kernel<<<(unsigned)((ntot + 255) / 256), 256, 0, stream>>>(
        proposals, features, Abf);

    // FC1: (2000 x 12544) @ (12544 x 1024) + b1, relu, bf16 out
    dim3 g1((MROWS + TM - 1) / TM, HID / TN);
    gemm_bf16_wmma_kernel<<<g1, 256, 0, stream>>>(
        Abf, KDIM, W1T, KDIM, b1, H1, HID, MROWS, HID, KDIM, 1, 1);
    // FC2
    gemm_bf16_wmma_kernel<<<g1, 256, 0, stream>>>(
        H1, HID, W2T, HID, b2, H2, HID, MROWS, HID, HID, 1, 1);
    // heads (cls+reg fused), f32 out
    dim3 g3((MROWS + TM - 1) / TM, (NHEAD + TN - 1) / TN);
    gemm_bf16_wmma_kernel<<<g3, 256, 0, stream>>>(
        H2, HID, WhT, HID, bh, Ch, NHEAD, MROWS, NHEAD, HID, 0, 0);

    // postprocess
    prep_kernel<<<MROWS, 128, 0, stream>>>(proposals, Ch, boxesW, scoresW, s0W, areasW);
    nms_kernel<<<BATCH, 1024, 0, stream>>>(boxesW, scoresW, s0W, areasW, scurW, out);
}
